// MultiHeadAttention_83107617178388
// MI455X (gfx1250) — matrix operand order repaired and checked by reading, not compiled, not run on hardware
//
#include <hip/hip_runtime.h>

#define DEV_INLINE __device__ __forceinline__

typedef __attribute__((ext_vector_type(16))) _Float16 v16h;
typedef __attribute__((ext_vector_type(8)))  _Float16 v8h;
typedef __attribute__((ext_vector_type(8)))  float    v8f;
typedef __attribute__((ext_vector_type(4)))  float    v4f;
typedef __attribute__((ext_vector_type(4)))  unsigned int u32x4;
typedef __attribute__((ext_vector_type(8)))  int      i32x8;
typedef __attribute__((ext_vector_type(4)))  int      i32x4;

constexpr int Bn = 4, Tn = 2048, En = 512, Hn = 8, Dn = 64;
constexpr int HD = Hn * Dn;          // 512
constexpr int N_QKV = 3 * HD;        // 1536

#if __has_builtin(__builtin_amdgcn_tensor_load_to_lds) && \
    __has_builtin(__builtin_amdgcn_s_wait_tensorcnt)
#define USE_TDM 1
#else
#define USE_TDM 0
#endif

// ---- WMMA fragment loaders (CDNA5 16x16x32 f16 layouts, wave32) ----
// A (16xK): lane row M=lane&15, g=lane>>4; e0..7 -> K=k0+8g+e, e8..15 -> K=k0+16+8g+(e-8)
DEV_INLINE v16h load_a_f16(const _Float16* rp, int k0, int g) {
  v8h x0 = *(const v8h*)(rp + k0 + 8 * g);
  v8h x1 = *(const v8h*)(rp + k0 + 16 + 8 * g);
  v16h r;
#pragma unroll
  for (int i = 0; i < 8; ++i) { r[i] = x0[i]; r[8 + i] = x1[i]; }
  return r;
}

// B (Kx16): lane col N=lane&15; e0..7 -> K=k0+8g+e, e8..15 -> K=k0+16+8g+(e-8)
DEV_INLINE v16h load_b_f16(const _Float16* rp, int k0, int g) {
  v8h x0 = *(const v8h*)(rp + k0 + 8 * g);
  v8h x1 = *(const v8h*)(rp + k0 + 16 + 8 * g);
  v16h r;
#pragma unroll
  for (int i = 0; i < 8; ++i) { r[i] = x0[i]; r[8 + i] = x1[i]; }
  return r;
}

DEV_INLINE v8f wmma16x16x32(v16h a, v16h b, v8f c) {
  return __builtin_amdgcn_wmma_f32_16x16x32_f16(false, a, false, b, (short)0, c,
                                                false, false);
}

// ---------------------------------------------------------------------------
// Tensor Data Mover helpers (D# descriptor per CDNA5 ISA).
// group0: [1:0]=count, [63:32]=lds_addr, [120:64]=global_addr, [127:126]=type(2)
// group1: [17:16]=data_size, [79:48]=tensor_dim0, [111:80]=tensor_dim1,
//         [127:112]=tile_dim0, [143:128]=tile_dim1, [207:160]=tensor_dim0_stride
// This toolchain's builtin is the 6-arg form:
//   (u32x4 g0, i32x8 g1, i32x4 g2, i32x4 g3, i32x8 extra, i32 cpol)
// ---------------------------------------------------------------------------
#if USE_TDM
DEV_INLINE void tdm_load_tile(unsigned int lds_off, const void* gptr,
                              unsigned int tile_d0, unsigned int tile_d1,
                              unsigned long long stride0_elems) {
  unsigned long long ga = (unsigned long long)(uintptr_t)gptr;
  const unsigned int td0 = 1u << 20, td1 = 1u << 20;  // big dims: no OOB clamp
  u32x4 g0;
  g0[0] = 1u;                                           // count=1, user mode
  g0[1] = lds_off;                                      // lds byte address
  g0[2] = (unsigned int)ga;                             // global_addr[31:0]
  g0[3] = (unsigned int)((ga >> 32) & 0x01FFFFFFu) | (2u << 30);  // type=2
  i32x8 g1;
  g1[0] = (int)(1u << 16);                              // data_size=1 (2 bytes)
  g1[1] = (int)((td0 & 0xFFFFu) << 16);                 // tensor_dim0[15:0]
  g1[2] = (int)((td0 >> 16) | ((td1 & 0xFFFFu) << 16)); // td0[31:16] | td1[15:0]
  g1[3] = (int)((td1 >> 16) | (tile_d0 << 16));         // td1[31:16] | tile_dim0
  g1[4] = (int)(tile_d1 & 0xFFFFu);                     // tile_dim1 (0 = 1-D)
  g1[5] = (int)(unsigned int)(stride0_elems & 0xFFFFFFFFull);
  g1[6] = (int)(unsigned int)((stride0_elems >> 32) & 0xFFFFull);
  g1[7] = 0;
  i32x4 z4 = {0, 0, 0, 0};
  i32x8 z8 = {0, 0, 0, 0, 0, 0, 0, 0};
  __builtin_amdgcn_tensor_load_to_lds(g0, g1, z4, z4, z8, 0);
}
#endif

// ---------------------------------------------------------------------------
// Kernel 0: bulk fp32 -> fp16 convert (x, W_qkv, W_0 each converted once).
// ---------------------------------------------------------------------------
__global__ __launch_bounds__(256) void cvt_f16_kernel(
    const float* __restrict__ src, _Float16* __restrict__ dst, int n8) {
  int i = blockIdx.x * blockDim.x + threadIdx.x;
  if (i >= n8) return;
  const v4f* s = (const v4f*)src + (size_t)i * 2;
  v4f a = s[0], b = s[1];
  v8h h;
#pragma unroll
  for (int k = 0; k < 4; ++k) { h[k] = (_Float16)a[k]; h[4 + k] = (_Float16)b[k]; }
  *((v8h*)dst + i) = h;
}

// ---------------------------------------------------------------------------
// Kernel 1: qkv = x @ Wqkv^T (8192x1536x512, all-fp16 operands); scatter into
// Q(pre-scaled by D^-0.5), K [B,H,T,D] and V^T [B,H,D,T], fp16.
// One wave computes a 16x64 tile; B fragments kept live to clause loads.
// ---------------------------------------------------------------------------
__global__ __launch_bounds__(256) void qkv_proj_kernel(
    const _Float16* __restrict__ xh, const _Float16* __restrict__ Wh,
    _Float16* __restrict__ Qh, _Float16* __restrict__ Kh,
    _Float16* __restrict__ Vt)
{
  const int lane = threadIdx.x & 31;
  const int wave = blockIdx.x * (blockDim.x >> 5) + (threadIdx.x >> 5);
  const int NG = N_QKV / 64;                 // 24 column-groups
  const int mt = wave / NG;
  const int ng = wave % NG;
  if (mt >= (Bn * Tn) / 16) return;
  const int m0 = mt * 16, n0 = ng * 64;
  const int g = lane >> 4, ln = lane & 15;

  const _Float16* arow = xh + (size_t)(m0 + ln) * En;
  const _Float16* brow0 = Wh + (size_t)(n0 + ln) * En;
  const _Float16* brow1 = brow0 + (size_t)16 * En;
  const _Float16* brow2 = brow0 + (size_t)32 * En;
  const _Float16* brow3 = brow0 + (size_t)48 * En;

  v8f acc0 = {}, acc1 = {}, acc2 = {}, acc3 = {};
  for (int k0 = 0; k0 < En; k0 += 32) {
    v16h a  = load_a_f16(arow, k0, g);
    v16h b0 = load_b_f16(brow0, k0, g);
    v16h b1 = load_b_f16(brow1, k0, g);
    v16h b2 = load_b_f16(brow2, k0, g);
    v16h b3 = load_b_f16(brow3, k0, g);
    acc0 = wmma16x16x32(a, b0, acc0);
    acc1 = wmma16x16x32(a, b1, acc1);
    acc2 = wmma16x16x32(a, b2, acc2);
    acc3 = wmma16x16x32(a, b3, acc3);
  }

  // reference reshape: n = h*(3*D) + d*3 + c  (c: 0=q, 1=k, 2=v)
  const float qscale = 0.125f;  // D^-0.5
  v8f accs[4] = {acc0, acc1, acc2, acc3};
#pragma unroll
  for (int t = 0; t < 4; ++t) {
#pragma unroll
    for (int r = 0; r < 8; ++r) {
      const int m = m0 + r + 8 * g;
      const int n = n0 + t * 16 + ln;
      const int b_ = m / Tn, tt = m % Tn;
      const int h = n / (3 * Dn);
      const int rem = n - h * (3 * Dn);
      const int d = rem / 3, c = rem - d * 3;
      const float v = accs[t][r];
      const size_t bh = (size_t)b_ * Hn + h;
      if (c == 0)       Qh[(bh * Tn + tt) * Dn + d] = (_Float16)(v * qscale);
      else if (c == 1)  Kh[(bh * Tn + tt) * Dn + d] = (_Float16)v;
      else              Vt[(bh * Dn + d) * Tn + tt] = (_Float16)v;
    }
  }
}

// ---------------------------------------------------------------------------
// Kernel 2: flash attention. Block = 8 waves on 128 consecutive query rows of
// one (b,h). K/V tiles (32 keys/step) are staged into double-buffered LDS by
// the Tensor Data Mover (fallback: cooperative vector loads) and shared by all
// 8 waves. Online softmax in fp32; P converted through per-wave LDS.
// ---------------------------------------------------------------------------
__global__ __launch_bounds__(256) void flash_attn_kernel(
    const _Float16* __restrict__ Qh, const _Float16* __restrict__ Kh,
    const _Float16* __restrict__ Vt, _Float16* __restrict__ Ah)
{
  __shared__ _Float16 kbuf[2][32 * 64];      // 32 keys x 64 d      (4KB each)
  __shared__ _Float16 vbuf[2][64 * 32];      // 64 d    x 32 keys   (4KB each)
  __shared__ _Float16 pstage[8][16 * 32];    // per-wave P tile     (1KB each)

  const int lane  = threadIdx.x & 31;
  const int wslot = threadIdx.x >> 5;
  const int bh = blockIdx.x >> 4;            // 16 q-blocks (128 rows) per (b,h)
  const int qb = blockIdx.x & 15;
  const int qt = qb * 8 + wslot;             // this wave's 16-row q tile
  const int g = lane >> 4, ln = lane & 15;
  const bool wave0 = (__builtin_amdgcn_readfirstlane((int)(threadIdx.x >> 5)) == 0);

  const _Float16* qbase = Qh + ((size_t)bh * Tn + qt * 16) * Dn;
  const _Float16* kbase = Kh + (size_t)bh * Tn * Dn;
  const _Float16* vbase = Vt + (size_t)bh * Dn * Tn;

  const v16h qa0 = load_a_f16(qbase + (size_t)ln * Dn, 0, g);
  const v16h qa1 = load_a_f16(qbase + (size_t)ln * Dn, 32, g);

  v8f o[4] = {};
  float mrow[8], lrow[8];
#pragma unroll
  for (int r = 0; r < 8; ++r) { mrow[r] = -1e30f; lrow[r] = 0.f; }
  _Float16* P = &pstage[wslot][0];

  // ---- staging helpers ----
  auto stage = [&](int buf, int j0) {
#if USE_TDM
    if (wave0) {
      // K tile: 32 rows x 64 halfs, rows contiguous -> 1-D 4KB copy
      tdm_load_tile((unsigned int)(uintptr_t)&kbuf[buf][0],
                    kbase + (size_t)j0 * Dn, 32 * 64, 0, 0);
      // V tile: 64 rows x 32 halfs, row stride = T elements -> 2-D copy
      tdm_load_tile((unsigned int)(uintptr_t)&vbuf[buf][0],
                    vbase + j0, 32, 64, (unsigned long long)Tn);
    }
#else
    const int tid = threadIdx.x;
    *(v8h*)&kbuf[buf][tid * 8] = *(const v8h*)(kbase + (size_t)j0 * Dn + tid * 8);
    const int vr = tid >> 2, vs = tid & 3;
    *(v8h*)&vbuf[buf][vr * 32 + vs * 8] =
        *(const v8h*)(vbase + (size_t)vr * Tn + j0 + vs * 8);
#endif
  };

  int buf = 0;
  stage(buf, 0);

  for (int j0 = 0; j0 < Tn; j0 += 32) {
#if USE_TDM
    if (wave0) __builtin_amdgcn_s_wait_tensorcnt(0);
#endif
    __syncthreads();                          // staged tile visible to block
    const int nbuf = buf ^ 1;
    if (j0 + 32 < Tn) stage(nbuf, j0 + 32);

    const _Float16* kb = &kbuf[buf][0];
    const _Float16* vb = &vbuf[buf][0];

    // scores: two 16x16 C tiles over 32 keys, K-dim = D = 64
    v16h kb00 = load_b_f16(kb + (size_t)ln * 64, 0, g);
    v16h kb01 = load_b_f16(kb + (size_t)ln * 64, 32, g);
    v16h kb10 = load_b_f16(kb + (size_t)(16 + ln) * 64, 0, g);
    v16h kb11 = load_b_f16(kb + (size_t)(16 + ln) * 64, 32, g);
    v8f s0 = {}, s1 = {};
    s0 = wmma16x16x32(qa0, kb00, s0);
    s0 = wmma16x16x32(qa1, kb01, s0);
    s1 = wmma16x16x32(qa0, kb10, s1);
    s1 = wmma16x16x32(qa1, kb11, s1);

    // online softmax; row r lives in 16-lane halves of the wave
#pragma unroll
    for (int r = 0; r < 8; ++r) {
      float mx = fmaxf(s0[r], s1[r]);
      mx = fmaxf(mx, __shfl_xor(mx, 1, 32));
      mx = fmaxf(mx, __shfl_xor(mx, 2, 32));
      mx = fmaxf(mx, __shfl_xor(mx, 4, 32));
      mx = fmaxf(mx, __shfl_xor(mx, 8, 32));
      const float mnew = fmaxf(mrow[r], mx);
      const float alpha = __expf(mrow[r] - mnew);
      const float p0 = __expf(s0[r] - mnew);
      const float p1 = __expf(s1[r] - mnew);
      float ps = p0 + p1;
      ps += __shfl_xor(ps, 1, 32);
      ps += __shfl_xor(ps, 2, 32);
      ps += __shfl_xor(ps, 4, 32);
      ps += __shfl_xor(ps, 8, 32);
      lrow[r] = lrow[r] * alpha + ps;
      mrow[r] = mnew;
#pragma unroll
      for (int t = 0; t < 4; ++t) o[t][r] *= alpha;
      const int row = r + 8 * g;
      P[row * 32 + ln]      = (_Float16)p0;
      P[row * 32 + 16 + ln] = (_Float16)p1;
    }
    __builtin_amdgcn_wave_barrier();
    const v16h pa = load_a_f16(P + ln * 32, 0, g);   // P as A fragment
    __builtin_amdgcn_wave_barrier();

    // O += P @ V from LDS V^T tile (rows are d, contiguous in j)
    v16h vb0 = load_b_f16(vb + (size_t)(0 * 16 + ln) * 32, 0, g);
    v16h vb1 = load_b_f16(vb + (size_t)(1 * 16 + ln) * 32, 0, g);
    v16h vb2 = load_b_f16(vb + (size_t)(2 * 16 + ln) * 32, 0, g);
    v16h vb3 = load_b_f16(vb + (size_t)(3 * 16 + ln) * 32, 0, g);
    o[0] = wmma16x16x32(pa, vb0, o[0]);
    o[1] = wmma16x16x32(pa, vb1, o[1]);
    o[2] = wmma16x16x32(pa, vb2, o[2]);
    o[3] = wmma16x16x32(pa, vb3, o[3]);

    __syncthreads();                          // block done with buf
    buf = nbuf;
  }

  const int b_ = bh / Hn, h = bh % Hn;
#pragma unroll
  for (int r = 0; r < 8; ++r) {
    const float inv = 1.0f / lrow[r];
    const int tg = qt * 16 + r + 8 * g;
    _Float16* dst = Ah + (size_t)(b_ * Tn + tg) * HD + h * Dn;
#pragma unroll
    for (int t = 0; t < 4; ++t) dst[t * 16 + ln] = (_Float16)(o[t][r] * inv);
  }
}

// ---------------------------------------------------------------------------
// Kernel 3: out = attn_out @ W0^T (8192x512x512), fp16 operands, fp32 result.
// ---------------------------------------------------------------------------
__global__ __launch_bounds__(256) void out_proj_kernel(
    const _Float16* __restrict__ Ah, const _Float16* __restrict__ W0h,
    float* __restrict__ out)
{
  const int lane = threadIdx.x & 31;
  const int wave = blockIdx.x * (blockDim.x >> 5) + (threadIdx.x >> 5);
  const int NG = En / 64;                    // 8 column-groups
  const int mt = wave / NG;
  const int ng = wave % NG;
  if (mt >= (Bn * Tn) / 16) return;
  const int m0 = mt * 16, n0 = ng * 64;
  const int g = lane >> 4, ln = lane & 15;

  const _Float16* arow = Ah + (size_t)(m0 + ln) * HD;
  const _Float16* brow0 = W0h + (size_t)(n0 + ln) * HD;
  const _Float16* brow1 = brow0 + (size_t)16 * HD;
  const _Float16* brow2 = brow0 + (size_t)32 * HD;
  const _Float16* brow3 = brow0 + (size_t)48 * HD;

  v8f acc0 = {}, acc1 = {}, acc2 = {}, acc3 = {};
  for (int k0 = 0; k0 < HD; k0 += 32) {
    v16h a  = load_a_f16(arow, k0, g);
    v16h b0 = load_b_f16(brow0, k0, g);
    v16h b1 = load_b_f16(brow1, k0, g);
    v16h b2 = load_b_f16(brow2, k0, g);
    v16h b3 = load_b_f16(brow3, k0, g);
    acc0 = wmma16x16x32(a, b0, acc0);
    acc1 = wmma16x16x32(a, b1, acc1);
    acc2 = wmma16x16x32(a, b2, acc2);
    acc3 = wmma16x16x32(a, b3, acc3);
  }
  v8f accs[4] = {acc0, acc1, acc2, acc3};
#pragma unroll
  for (int t = 0; t < 4; ++t) {
#pragma unroll
    for (int r = 0; r < 8; ++r) {
      const int m = m0 + r + 8 * g;
      const int n = n0 + t * 16 + ln;
      out[(size_t)m * En + n] = accs[t][r];
    }
  }
}

// ---------------------------------------------------------------------------
extern "C" void kernel_launch(void* const* d_in, const int* in_sizes, int n_in,
                              void* d_out, int out_size, void* d_ws, size_t ws_size,
                              hipStream_t stream) {
  (void)in_sizes; (void)n_in; (void)out_size; (void)ws_size;
  const float* x    = (const float*)d_in[0];
  const float* Wqkv = (const float*)d_in[1];
  const float* W0   = (const float*)d_in[2];
  float* out = (float*)d_out;

  const size_t n_x    = (size_t)Bn * Tn * En;       // 4,194,304
  const size_t n_wq   = (size_t)N_QKV * En;         //   786,432
  const size_t n_w0   = (size_t)En * HD;            //   262,144
  const size_t nper   = (size_t)Bn * Hn * Tn * Dn;  // 4,194,304

  _Float16* xh   = (_Float16*)d_ws;
  _Float16* Wh   = xh + n_x;
  _Float16* W0h  = Wh + n_wq;
  _Float16* Qh   = W0h + n_w0;
  _Float16* Kh   = Qh + nper;
  _Float16* Vt   = Kh + nper;
  _Float16* Ah   = Vt + nper;

  cvt_f16_kernel<<<(int)(n_x  / 8 / 256), 256, 0, stream>>>(x, xh, (int)(n_x / 8));
  cvt_f16_kernel<<<(int)(n_wq / 8 / 256), 256, 0, stream>>>(Wqkv, Wh, (int)(n_wq / 8));
  cvt_f16_kernel<<<(int)(n_w0 / 8 / 256), 256, 0, stream>>>(W0, W0h, (int)(n_w0 / 8));

  const int waves1 = ((Bn * Tn) / 16) * (N_QKV / 64);  // 12288
  qkv_proj_kernel<<<waves1 / 8, 256, 0, stream>>>(xh, Wh, Qh, Kh, Vt);

  flash_attn_kernel<<<Bn * Hn * (Tn / 128), 256, 0, stream>>>(Qh, Kh, Vt, Ah);

  const int waves3 = ((Bn * Tn) / 16) * (En / 64);     // 4096
  out_proj_kernel<<<waves3 / 8, 256, 0, stream>>>(Ah, W0h, out);
}
